// Evoformer_72765335929070
// MI455X (gfx1250) — matrix operand order repaired and checked by reading, not compiled, not run on hardware
//
#include <hip/hip_runtime.h>
#include <math.h>

// ---------------- problem constants ----------------
#define S_ 128
#define R_ 256
#define CM_ 256
#define CZ_ 128
#define H_ 8
#define C_ 32
#define PH_ 4
#define CT_ 128
#define SCALE_ 0.17677669529663687f   // 1/sqrt(32)

typedef _Float16 h16;
typedef __attribute__((ext_vector_type(16))) _Float16 v16h;
typedef __attribute__((ext_vector_type(8)))  _Float16 h8v;
typedef __attribute__((ext_vector_type(8)))  float    v8f;

// =====================================================================
// Strided batched GEMM, vector-path only (every call site guarantees
// K-contiguous operands):  C[b] = alpha * A[b] @ B[b]
//   A(m,k) at A + boffA(b) + m*rsA + k          (K contiguous)
//   B(k,n) at B + boffB(b) + n*csB + k          (K contiguous)
//   C(m,n) at C + boffC(b) + m*rsC + n*csC
//   boffX(b) = (b/bd)*oX1 + (b%bd)*oX2
// (16*MT)x(16*NT) macro-tile per wave32, K stepped by 32 with
// V_WMMA_F32_16X16X32_F16 (f16 in, f32 accumulate, f16 out).
// Fragments are exactly two b128 loads; rsC==1 stores one b128 per tile row.
// =====================================================================
struct GemmP {
  const h16* A; const h16* B; h16* C;
  long rsA, csB, rsC, csC;
  long oA1, oA2, oB1, oB2, oC1, oC2;
  int  bd, M, N, K, batches;
  float alpha;
};

__device__ __forceinline__ v16h load_a_frag(const h16* __restrict__ Ap,
                                            long rsA, int am, int hf) {
  // lane needs K = hf*8+0..7 and 16+hf*8+0..7 -> two b128 loads
  const h16* rp = Ap + (long)am * rsA + (hf << 3);
  h8v r0 = *(const h8v*)(rp);
  h8v r1 = *(const h8v*)(rp + 16);
  v16h a;
#pragma unroll
  for (int e = 0; e < 8; ++e) { a[e] = r0[e]; a[8 + e] = r1[e]; }
  return a;
}

__device__ __forceinline__ v16h load_b_frag(const h16* __restrict__ Bp,
                                            long csB, int n, int hf) {
  // lane needs K = hf*8+0..7 and 16+hf*8+0..7 -> two b128 loads
  const h16* cp = Bp + (long)n * csB + (hf << 3);
  h8v r0 = *(const h8v*)(cp);
  h8v r1 = *(const h8v*)(cp + 16);
  v16h b;
#pragma unroll
  for (int e = 0; e < 8; ++e) { b[e] = r0[e]; b[8 + e] = r1[e]; }
  return b;
}

template<int MT, int NT>
__global__ __launch_bounds__(256) void gemm_wmma(GemmP p) {
  const int wave = threadIdx.x >> 5;
  const int lane = threadIdx.x & 31;
  const int tpbN = p.N / (16 * NT);
  const int tpb  = (p.M / (16 * MT)) * tpbN;
  long tid = (long)blockIdx.x * 8 + wave;
  if (tid >= (long)tpb * (long)p.batches) return;  // wave-uniform: EXEC stays all-ones
  int b  = (int)(tid / (long)tpb);
  int t  = (int)(tid % (long)tpb);
  int mt = t / tpbN, nt = t % tpbN;
  int bq = b / p.bd, br = b % p.bd;

  const h16* Ak = p.A + (long)bq * p.oA1 + (long)br * p.oA2 + (long)mt * 16 * MT * p.rsA;
  const h16* Bk = p.B + (long)bq * p.oB1 + (long)br * p.oB2 + (long)nt * 16 * NT * p.csB;
  h16*       Cb = p.C + (long)bq * p.oC1 + (long)br * p.oC2 +
                  (long)mt * 16 * MT * p.rsC + (long)nt * 16 * NT * p.csC;

  const int am = lane & 15;   // A row (M) / B col (N) / C col (N)
  const int hf = lane >> 4;   // lane half

  v8f acc[MT][NT] = {};
  for (int k0 = 0; k0 < p.K; k0 += 32) {
    __builtin_prefetch(Ak + 32, 0, 3);   // global_prefetch_b8: next K slab
    __builtin_prefetch(Bk + 32, 0, 3);
    v16h af[MT];
#pragma unroll
    for (int i = 0; i < MT; ++i)
      af[i] = load_a_frag(Ak + (long)i * 16 * p.rsA, p.rsA, am, hf);
#pragma unroll
    for (int j = 0; j < NT; ++j) {
      v16h bf = load_b_frag(Bk + (long)j * 16 * p.csB, p.csB, am, hf);
#pragma unroll
      for (int i = 0; i < MT; ++i)
        acc[i][j] = __builtin_amdgcn_wmma_f32_16x16x32_f16(false, af[i], false, bf,
                                                           (short)0, acc[i][j], false, false);
    }
    Ak += 32;
    Bk += 32;
  }
  // C/D f32 16x16: VGPR r holds M = r + 8*half, N = lane%16
  const int cm = hf << 3;
#pragma unroll
  for (int i = 0; i < MT; ++i)
#pragma unroll
    for (int j = 0; j < NT; ++j) {
      if (p.rsC == 1) {       // M-contiguous layout: one b128 store of 8 f16
        h16* base = Cb + (long)i * 16 + cm + (long)(j * 16 + am) * p.csC;
        h8v st;
#pragma unroll
        for (int r = 0; r < 8; ++r) st[r] = (h16)(acc[i][j][r] * p.alpha);
        *(h8v*)base = st;
      } else {
        h16* Ct = Cb + (long)i * 16 * p.rsC + (long)j * 16 * p.csC;
#pragma unroll
        for (int r = 0; r < 8; ++r)
          Ct[(long)(cm + r) * p.rsC + (long)am * p.csC] = (h16)(acc[i][j][r] * p.alpha);
      }
    }
}

// =====================================================================
// LayerNorm over last dim (COLS = 128 or 256), one wave per row.
// Input f32 (masters) or f16 (intermediates); output f16.
// =====================================================================
template<int COLS, typename TIN>
__global__ __launch_bounds__(256) void ln_kernel(const TIN* __restrict__ x,
                                                 const float* __restrict__ w,
                                                 const float* __restrict__ bb,
                                                 h16* __restrict__ y, long rows) {
  const int NV = COLS / 32;
  long row = (long)blockIdx.x * 8 + (threadIdx.x >> 5);
  if (row >= rows) return;
  int l = threadIdx.x & 31;
  const TIN* xr = x + row * COLS;
  float v[NV];
  float s = 0.f;
#pragma unroll
  for (int i = 0; i < NV; ++i) { v[i] = (float)xr[l + 32 * i]; s += v[i]; }
#pragma unroll
  for (int o = 16; o >= 1; o >>= 1) s += __shfl_xor(s, o, 32);
  float mu = s * (1.f / COLS);
  float var = 0.f;
#pragma unroll
  for (int i = 0; i < NV; ++i) { float d = v[i] - mu; var += d * d; }
#pragma unroll
  for (int o = 16; o >= 1; o >>= 1) var += __shfl_xor(var, o, 32);
  float inv = rsqrtf(var * (1.f / COLS) + 1e-5f);
  h16* yr = y + row * COLS;
#pragma unroll
  for (int i = 0; i < NV; ++i)
    yr[l + 32 * i] = (h16)((v[i] - mu) * inv * w[l + 32 * i] + bb[l + 32 * i]);
}

// =====================================================================
// Softmax over last dim (L = 128 or 256), one wave per row, optional bias:
//   x[row, c] += bias[ ((row/d1)%m1)*s1 + ((row/d2)%m2)*s2 + c*cs ]
// =====================================================================
template<int L>
__global__ __launch_bounds__(256) void softmax_kernel(h16* __restrict__ x, long rows,
    const h16* __restrict__ bias, long d1, long m1, long s1,
    long d2, long m2, long s2, long cs) {
  const int NV = L / 32;
  long row = (long)blockIdx.x * 8 + (threadIdx.x >> 5);
  if (row >= rows) return;
  int l = threadIdx.x & 31;
  h16* xr = x + row * L;
  long boff = 0;
  if (bias) boff = ((row / d1) % m1) * s1 + ((row / d2) % m2) * s2;
  float v[NV];
  float mx = -3.4e38f;
#pragma unroll
  for (int i = 0; i < NV; ++i) {
    int c = l + 32 * i;
    float t = (float)xr[c];
    if (bias) t += (float)bias[boff + (long)c * cs];
    v[i] = t; mx = fmaxf(mx, t);
  }
#pragma unroll
  for (int o = 16; o >= 1; o >>= 1) mx = fmaxf(mx, __shfl_xor(mx, o, 32));
  float s = 0.f;
#pragma unroll
  for (int i = 0; i < NV; ++i) { v[i] = expf(v[i] - mx); s += v[i]; }
#pragma unroll
  for (int o = 16; o >= 1; o >>= 1) s += __shfl_xor(s, o, 32);
  float inv = 1.f / s;
#pragma unroll
  for (int i = 0; i < NV; ++i) xr[l + 32 * i] = (h16)(v[i] * inv);
}

// =====================================================================
// Tiny-N projection (N = 4 or 8): Y[Mr,N] = A[Mr,K] @ W[K,N]  (VALU)
// =====================================================================
template<typename TIN>
__global__ void small_nk(const TIN* __restrict__ A, const float* __restrict__ W,
                         h16* __restrict__ Y, long Mr, int K, int N) {
  long t = (long)blockIdx.x * blockDim.x + threadIdx.x;
  if (t >= Mr * (long)N) return;
  long r = t / N; int n = (int)(t % N);
  const TIN* a = A + r * K;
  float s = 0.f;
  for (int k = 0; k < K; ++k) s += (float)a[k] * W[(long)k * N + n];
  Y[t] = (h16)s;
}

// weight transpose + f32->f16: src[Kk,Nn] -> dst[Nn,Kk]
__global__ void transpose_kernel(const float* __restrict__ src, h16* __restrict__ dst,
                                 int Kk, int Nn) {
  long t = (long)blockIdx.x * 256 + threadIdx.x;
  if (t >= (long)Kk * Nn) return;
  int k = (int)(t / Nn), n = (int)(t % Nn);
  dst[(long)n * Kk + k] = (h16)src[t];
}

// gather V into K-contiguous layout: dst[(b*32+n)*K + k] = src[(b/bd)*g1 + (b%bd)*g2 + k*gk + n]
__global__ void gather_vt(const h16* __restrict__ src, h16* __restrict__ dst,
                          long total, int K, int bd, long g1, long g2, long gk) {
  long t = (long)blockIdx.x * 256 + threadIdx.x;
  if (t >= total) return;
  long k = t % K;
  long bn = t / K;
  long n = bn % 32;
  long b = bn / 32;
  dst[t] = src[(b / bd) * g1 + (b % bd) * g2 + k * gk + n];
}

// ---------------- elementwise helpers ----------------
__device__ __forceinline__ float sigm(float x) { return 1.f / (1.f + expf(-x)); }

__global__ void ew_sigmul(h16* g, const h16* o, long n) {              // g = sigmoid(g)*o
  long i = (long)blockIdx.x * 256 + threadIdx.x;
  if (i < n) g[i] = (h16)(sigm((float)g[i]) * (float)o[i]);
}
__global__ void ew_mulsig(const h16* a, const h16* b, h16* out, long n) { // out = a*sigmoid(b)
  long i = (long)blockIdx.x * 256 + threadIdx.x;
  if (i < n) out[i] = (h16)((float)a[i] * sigm((float)b[i]));
}
__global__ void ew_relu(h16* x, long n) {
  long i = (long)blockIdx.x * 256 + threadIdx.x;
  if (i < n) x[i] = (h16)fmaxf((float)x[i], 0.f);
}
__global__ void ew_add(float* y, const h16* x, long n) {               // master += delta
  long i = (long)blockIdx.x * 256 + threadIdx.x;
  if (i < n) y[i] += (float)x[i];
}
__global__ void ew_add3(const h16* a, const h16* b, float* out, long n) { // master = a + b
  long i = (long)blockIdx.x * 256 + threadIdx.x;
  if (i < n) out[i] = (float)a[i] + (float)b[i];
}
__global__ void ew_resgate(const h16* z, const h16* g, const h16* x, float* out, long n) {
  long i = (long)blockIdx.x * 256 + threadIdx.x;                       // master = z + sigmoid(g)*x
  if (i < n) out[i] = (float)z[i] + sigm((float)g[i]) * (float)x[i];
}

// =====================================================================
// host-side helpers
// =====================================================================
static void G(hipStream_t st, const h16* A, const h16* B, h16* Cc,
              int batches, int bd, int M, int N, int K,
              long rsA, long csB, long rsC, long csC,
              long oA1, long oA2, long oB1, long oB2, long oC1, long oC2, float alpha) {
  GemmP p{A, B, Cc, rsA, csB, rsC, csC,
          oA1, oA2, oB1, oB2, oC1, oC2, bd, M, N, K, batches, alpha};
  int MT = (M % 64 == 0) ? 4 : 2;
  int NT = (N % 64 == 0) ? 4 : 2;
  long total = (long)(M / (16 * MT)) * (N / (16 * NT)) * batches;
  int blocks = (int)((total + 7) / 8);
  if (MT == 4 && NT == 4)      gemm_wmma<4, 4><<<blocks, 256, 0, st>>>(p);
  else if (MT == 4)            gemm_wmma<4, 2><<<blocks, 256, 0, st>>>(p);
  else if (NT == 4)            gemm_wmma<2, 4><<<blocks, 256, 0, st>>>(p);
  else                         gemm_wmma<2, 2><<<blocks, 256, 0, st>>>(p);
}
// dense row-major GEMM with pre-transposed weight Bt[N,K]: C[M,N] = A[M,K] @ B
static void Gd(hipStream_t st, const h16* A, const h16* Bt, h16* Cc,
               int M, int N, int K, float alpha = 1.f) {
  G(st, A, Bt, Cc, 1, 1, M, N, K, K, K, N, 1, 0, 0, 0, 0, 0, 0, alpha);
}
static inline int eb(long n) { return (int)((n + 255) / 256); }

extern "C" void kernel_launch(void* const* d_in, const int* in_sizes, int n_in,
                              void* d_out, int out_size, void* d_ws, size_t ws_size,
                              hipStream_t stream) {
  (void)in_sizes; (void)n_in; (void)out_size; (void)ws_size;
  #define IN(i) ((const float*)d_in[(i)])

  const long EM = (long)S_ * R_ * CM_;   // 8388608 msa elements
  const long EP = (long)R_ * R_ * CZ_;   // 8388608 pair elements
  float* msa  = (float*)d_out;           // working msa lives in d_out (f32)
  float* pair = msa + EM;                // working pair lives in d_out (f32)

  // workspace layout (f16 elements); ~128M f16 (~256 MB)
  h16* mln  = (h16*)d_ws;                // 8388608   ln output / z
  h16* gate = mln  + 8388608;            // 8388608   gates / right / a-proj
  h16* obuf = gate + 8388608;            // 8388608   attention out / left / deltas
  h16* qkv  = obuf + 8388608;            // 25165824  qkv / tri-mult temporaries
  h16* lg   = qkv  + 25165824;           // 67108864  logits / outer / ffn hidden
  h16* sb   = lg   + 67108864;           // 524288    small bias projections
  h16* vt   = sb   + 524288;             // 8388608   K-contiguous V for PV gemms
  h16* wt   = vt   + 8388608;            // ~1.9M     transposed dense weights
  h16* x1 = qkv + 8388608;               // tri-mult temp 1
  h16* x2 = qkv + 16777216;              // tri-mult temp 2

  hipMemcpyAsync(msa,  IN(0), EM * sizeof(float), hipMemcpyDeviceToDevice, stream);
  hipMemcpyAsync(pair, IN(1), EP * sizeof(float), hipMemcpyDeviceToDevice, stream);

  // ---- pre-transpose all dense weights (f32 W[K,N] -> f16 Wt[N,K]) ----
  h16* cur = wt;
  auto T = [&](int idx, int Kk, int Nn) -> const h16* {
    h16* dst = cur; cur += (long)Kk * Nn;
    transpose_kernel<<<eb((long)Kk * Nn), 256, 0, stream>>>(IN(idx), dst, Kk, Nn);
    return dst;
  };
  const h16* t_rgate = T(4, 256, 256);   const h16* t_rqkv = T(5, 256, 768);
  const h16* t_rout  = T(6, 256, 256);
  const h16* t_cgate = T(10, 256, 256);  const h16* t_cqkv = T(11, 256, 768);
  const h16* t_cout  = T(12, 256, 256);
  const h16* t_mtp1  = T(15, 256, 1024); const h16* t_mtp2 = T(16, 1024, 256);
  const h16* t_op1   = T(19, 256, 32);   const h16* t_op2  = T(20, 256, 32);
  const h16* t_oout  = T(21, 1024, 128);
  const h16* t_tm[2][6];
  for (int pass = 0; pass < 2; ++pass) {
    int w0 = (pass == 0) ? 22 : 32;
    for (int k = 0; k < 6; ++k) t_tm[pass][k] = T(w0 + 4 + k, 128, 128); // p1..p5,p6
  }
  const h16* t_ta[2][3];
  for (int pass = 0; pass < 2; ++pass) {
    int w0 = (pass == 0) ? 42 : 48;
    t_ta[pass][0] = T(w0 + 2, 128, 128);   // gate
    t_ta[pass][1] = T(w0 + 3, 128, 384);   // qkv
    t_ta[pass][2] = T(w0 + 4, 128, 128);   // out
  }
  const h16* t_ptp1 = T(56, 128, 512);
  const h16* t_ptp2 = T(57, 512, 128);

  // ================= MSA row attention with pair bias =================
  ln_kernel<256, float><<<4096, 256, 0, stream>>>(msa, IN(2), IN(3), mln, 32768);
  Gd(stream, mln, t_rqkv, qkv, 32768, 768, 256);                      // qkv = m @ r_qkv
  Gd(stream, mln, t_rgate, gate, 32768, 256, 256);                    // gate = m @ r_gate
  small_nk<float><<<eb(65536L * 8), 256, 0, stream>>>(pair, IN(7), sb, 65536, 128, 8);
  // logits[s,h,i,j] = SCALE * q . k    (batch (s,h), bd=H)
  G(stream, qkv, qkv + 256, lg, 1024, 8, 256, 256, 32,
    768, 768, 256, 1,
    196608, 32, 196608, 32, 524288, 65536, SCALE_);
  softmax_kernel<256><<<32768, 256, 0, stream>>>(lg, 262144, sb,
    256, 8, 1, 1, 256, 2048, 8);                                      // += bias[h,i,j]
  // o[s,i,h,c] = P @ v  with K-contiguous v_t
  gather_vt<<<eb(8388608L), 256, 0, stream>>>(qkv + 512, vt, 8388608, 256, 8, 196608, 32, 768);
  G(stream, lg, vt, obuf, 1024, 8, 256, 32, 256,
    256, 256, 256, 1,
    524288, 65536, 65536, 8192, 65536, 32, 1.f);
  ew_sigmul<<<eb(EM), 256, 0, stream>>>(gate, obuf, EM);
  Gd(stream, gate, t_rout, obuf, 32768, 256, 256);                    // delta = (g*o) @ r_out
  ew_add<<<eb(EM), 256, 0, stream>>>(msa, obuf, EM);

  // ================= MSA column attention =================
  ln_kernel<256, float><<<4096, 256, 0, stream>>>(msa, IN(8), IN(9), mln, 32768);
  Gd(stream, mln, t_cqkv, qkv, 32768, 768, 256);
  Gd(stream, mln, t_cgate, gate, 32768, 256, 256);
  // logits[h,r,i,j] over sequence dim (batch (h,r), bd=R)
  G(stream, qkv, qkv + 256, lg, 2048, 256, 128, 128, 32,
    196608, 196608, 128, 1,
    32, 768, 32, 768, 4194304, 16384, SCALE_);
  softmax_kernel<128><<<32768, 256, 0, stream>>>(lg, 262144, nullptr, 1, 1, 0, 1, 1, 0, 0);
  gather_vt<<<eb(8388608L), 256, 0, stream>>>(qkv + 512, vt, 8388608, 128, 256, 32, 768, 196608);
  G(stream, lg, vt, obuf, 2048, 256, 128, 32, 128,
    128, 128, 65536, 1,
    4194304, 16384, 1048576, 4096, 32, 256, 1.f);
  ew_sigmul<<<eb(EM), 256, 0, stream>>>(gate, obuf, EM);
  Gd(stream, gate, t_cout, obuf, 32768, 256, 256);
  ew_add<<<eb(EM), 256, 0, stream>>>(msa, obuf, EM);

  // ================= MSA transition =================
  ln_kernel<256, float><<<4096, 256, 0, stream>>>(msa, IN(13), IN(14), mln, 32768);
  Gd(stream, mln, t_mtp1, lg, 32768, 1024, 256);
  ew_relu<<<eb(33554432L), 256, 0, stream>>>(lg, 33554432);
  Gd(stream, lg, t_mtp2, obuf, 32768, 256, 1024);
  ew_add<<<eb(EM), 256, 0, stream>>>(msa, obuf, EM);

  // ================= Outer product mean =================
  ln_kernel<256, float><<<4096, 256, 0, stream>>>(msa, IN(17), IN(18), mln, 32768);
  // a_t/b_t in [i, c, s] layout (S-contiguous) via batched store (batch s)
  G(stream, mln, t_op1, gate, 128, 1, 256, 32, 256,
    256, 256, 4096, 128, 65536, 0, 0, 0, 1, 0, 1.f);
  G(stream, mln, t_op2, obuf, 128, 1, 256, 32, 256,
    256, 256, 4096, 128, 65536, 0, 0, 0, 1, 0, 1.f);
  // outer[i,j,c,d] = (1/S) sum_s a_t[i,c,s] b_t[j,d,s]   (batch (i,j), bd=R)
  G(stream, gate, obuf, lg, 65536, 256, 32, 32, 128,
    128, 128, 32, 1,
    4096, 0, 0, 4096, 262144, 1024, 1.f / S_);
  Gd(stream, lg, t_oout, obuf, 65536, 128, 1024);                     // delta = outer @ o_out
  ew_add<<<eb(EP), 256, 0, stream>>>(pair, obuf, EP);

  // ================= Triangle multiplicative updates =================
  for (int pass = 0; pass < 2; ++pass) {
    int w0 = (pass == 0) ? 22 : 32;   // tmo_* or tmi_*
    ln_kernel<128, float><<<8192, 256, 0, stream>>>(pair, IN(w0), IN(w0 + 1), mln, 65536); // z
    // projections stored channel-major so both einsum passes hit vector paths:
    //   pass0: [c, a, b] (contract over b=k);  pass1: [c, b, a] (contract over a=k)
    auto Gc = [&](const h16* tw, h16* dst) {
      if (pass == 0)
        G(stream, mln, tw, dst, 1, 1, 65536, 128, 128,
          128, 128, 1, 65536, 0, 0, 0, 0, 0, 0, 1.f);
      else
        G(stream, mln, tw, dst, 256, 1, 256, 128, 128,
          128, 128, 256, 65536, 32768, 0, 0, 0, 1, 0, 1.f);
    };
    Gc(t_tm[pass][0], x1); Gc(t_tm[pass][1], x2);
    ew_mulsig<<<eb(EP), 256, 0, stream>>>(x1, x2, obuf, EP);          // left  (channel-major)
    Gc(t_tm[pass][2], x1); Gc(t_tm[pass][3], x2);
    ew_mulsig<<<eb(EP), 256, 0, stream>>>(x1, x2, gate, EP);          // right (channel-major)
    Gd(stream, mln, t_tm[pass][5], qkv, 65536, 128, 128);             // g = z @ p6 (row-major)
    // x[i,j,c] = sum_k L[i,k,c] R[j,k,c]  (both passes same strides by construction)
    G(stream, obuf, gate, x1, 128, 1, 256, 256, 256,
      256, 256, 32768, 128,
      65536, 0, 65536, 0, 1, 0, 1.f);
    ln_kernel<128, h16><<<8192, 256, 0, stream>>>(x1, IN(w0 + 2), IN(w0 + 3), x1, 65536);
    Gd(stream, x1, t_tm[pass][4], x2, 65536, 128, 128);               // x @ p5
    ew_resgate<<<eb(EP), 256, 0, stream>>>(mln, qkv, x2, pair, EP);   // pair = z + sig(g)*x
  }

  // ================= Triangle attentions =================
  for (int pass = 0; pass < 2; ++pass) {
    int w0 = (pass == 0) ? 42 : 48;   // tas_* or tae_*
    ln_kernel<128, float><<<8192, 256, 0, stream>>>(pair, IN(w0), IN(w0 + 1), mln, 65536); // z
    Gd(stream, mln, t_ta[pass][1], qkv, 65536, 384, 128);             // qkv = z @ *_qkv
    Gd(stream, mln, t_ta[pass][0], gate, 65536, 128, 128);            // gate
    small_nk<h16><<<eb(65536L * 4), 256, 0, stream>>>(mln, IN(w0 + 5), sb, 65536, 128, 4);
    if (pass == 0) {
      // start: logits[h,i,j,k] = q_ij . k_ik   (batch (h,i), bd=R)
      G(stream, qkv, qkv + 128, lg, 1024, 256, 256, 256, 32,
        384, 384, 256, 1,
        32, 98304, 32, 98304, 16777216, 65536, SCALE_);
      softmax_kernel<256><<<32768, 256, 0, stream>>>(lg, 262144, sb,
        65536, 4, 1, 1, 256, 1024, 4);                                // += b[h,j,k]
      gather_vt<<<eb(8388608L), 256, 0, stream>>>(qkv + 256, vt, 8388608, 256, 256, 32, 98304, 384);
      G(stream, lg, vt, obuf, 1024, 256, 256, 32, 256,
        256, 256, 128, 1,
        16777216, 65536, 2097152, 8192, 32, 32768, 1.f);              // o[i,j,h,c]
    } else {
      // end: logits[h,i,j,k] = q_ij . k_kj   (batch (h,j), bd=R)
      G(stream, qkv, qkv + 128, lg, 1024, 256, 256, 256, 32,
        98304, 98304, 65536, 1,
        32, 384, 32, 384, 16777216, 256, SCALE_);
      softmax_kernel<256><<<32768, 256, 0, stream>>>(lg, 262144, sb,
        65536, 4, 1, 256, 256, 4, 1024);                              // += b[h,k,i]
      gather_vt<<<eb(8388608L), 256, 0, stream>>>(qkv + 256, vt, 8388608, 256, 256, 32, 384, 98304);
      G(stream, lg, vt, obuf, 1024, 256, 256, 32, 256,
        65536, 256, 32768, 1,
        16777216, 256, 2097152, 8192, 32, 128, 1.f);                  // o[i,j,h,c]
    }
    ew_sigmul<<<eb(EP), 256, 0, stream>>>(gate, obuf, EP);
    Gd(stream, gate, t_ta[pass][2], qkv, 65536, 128, 128);            // delta = (g*o) @ *_out
    ew_add3<<<eb(EP), 256, 0, stream>>>(mln, qkv, pair, EP);          // pair = z + delta
  }

  // ================= Pair transition =================
  ln_kernel<128, float><<<8192, 256, 0, stream>>>(pair, IN(54), IN(55), mln, 65536);
  Gd(stream, mln, t_ptp1, lg, 65536, 512, 128);
  ew_relu<<<eb(33554432L), 256, 0, stream>>>(lg, 33554432);
  Gd(stream, lg, t_ptp2, obuf, 65536, 128, 512);
  ew_add<<<eb(EP), 256, 0, stream>>>(pair, obuf, EP);
}
